// DifferentiableFrameSelector_83726092468245
// MI455X (gfx1250) — matrix operand order repaired and checked by reading, not compiled, not run on hardware
//
#include <hip/hip_runtime.h>
#include <stdint.h>

#define T_LEN 16384
#define D_IN  2048
#define H_GRU 256
#define N_COMB 1536   // 3H forward + 3H backward
#define NF 8
#define NC 2

typedef __attribute__((ext_vector_type(16))) __bf16 v16bf;
typedef __attribute__((ext_vector_type(8)))  float  v8f;

union Frag { uint4 q[2]; v16bf v; };

__device__ __forceinline__ unsigned short f2bf(float f){
  unsigned u = __float_as_uint(f);
  u += 0x7fffu + ((u >> 16) & 1u);          // round-to-nearest-even
  return (unsigned short)(u >> 16);
}

__device__ __forceinline__ float sigm(float x){ return 1.f / (1.f + __expf(-x)); }

// LDS byte offset of a pointer into __shared__ memory: generic->LDS is a truncation on gfx1250.
__device__ __forceinline__ unsigned lds_off(const void* p){
  return (unsigned)(uintptr_t)p;
}

// async global -> LDS copy of 16 bytes (per lane), tracked by ASYNCcnt
__device__ __forceinline__ void async_copy_b128(unsigned lds_dst, const void* gsrc){
  asm volatile("global_load_async_to_lds_b128 %0, %1, off"
               :: "v"(lds_dst), "v"((unsigned long long)(uintptr_t)gsrc)
               : "memory");
}
__device__ __forceinline__ void wait_async0(){
  asm volatile("s_wait_asynccnt 0" ::: "memory");
}

// ---------------------------------------------------------------- stage 0: pack
__global__ void k_cvt_bf16(const float* __restrict__ src, unsigned short* __restrict__ dst, int n){
  int i = blockIdx.x * blockDim.x + threadIdx.x;
  int stride = gridDim.x * blockDim.x;
  for (; i < n; i += stride) dst[i] = f2bf(src[i]);
}

__global__ void k_pack_w(const float* __restrict__ Wf, const float* __restrict__ Wb,
                         unsigned short* __restrict__ Bt){
  int i = blockIdx.x * blockDim.x + threadIdx.x;
  int stride = gridDim.x * blockDim.x;
  const int total = N_COMB * D_IN;
  for (; i < total; i += stride){
    int n = i / D_IN, k = i - n * D_IN;
    float v = (n < 768) ? Wf[n * D_IN + k] : Wb[(n - 768) * D_IN + k];
    Bt[i] = f2bf(v);                         // Bt[n][k] row-major == W_ih layout
  }
}

__global__ void k_pack_bias(const float* __restrict__ bf, const float* __restrict__ bb,
                            float* __restrict__ out){
  int i = blockIdx.x * blockDim.x + threadIdx.x;
  if (i < N_COMB) out[i] = (i < 768) ? bf[i] : bb[i - 768];
}

// ------------------------------------------------- stage 1: WMMA GEMM gx = A * B^T + bias
// A: (T, D) bf16 row-major. B: (N_COMB, D) bf16 row-major (acts as B^T of a (D, N) matrix).
// Block tile 64(M) x 128(N), 8 waves: wave (wm, wn) -> rows wm*16, cols wn*64 (four 16x16 tiles).
// Tiles staged to LDS with CDNA5 async global->LDS copies (ASYNCcnt).
__global__ void __launch_bounds__(256, 2)
k_gemm_gx(const unsigned short* __restrict__ A, const unsigned short* __restrict__ B,
          const float* __restrict__ bias, float* __restrict__ gx){
  __shared__ __align__(16) unsigned short As[64 * 40];    // stride 40 -> 16B aligned rows, conflict-free
  __shared__ __align__(16) unsigned short Bs[128 * 40];
  const int m0 = blockIdx.y * 64;
  const int n0 = blockIdx.x * 128;
  const int tid  = threadIdx.x;
  const int lane = tid & 31;
  const int wave = tid >> 5;
  const int wm = wave & 3;
  const int wn = wave >> 2;
  const int lrow = tid >> 2;            // 0..63
  const int lcol = (tid & 3) * 8;       // 0,8,16,24

  v8f acc0 = {0.f,0.f,0.f,0.f,0.f,0.f,0.f,0.f};
  v8f acc1 = acc0, acc2 = acc0, acc3 = acc0;

  // per-thread async copy destinations (constant across K loop)
  const unsigned a_dst  = lds_off(&As[lrow * 40 + lcol]);
  const unsigned b_dst0 = lds_off(&Bs[lrow * 40 + lcol]);
  const unsigned b_dst1 = lds_off(&Bs[(lrow + 64) * 40 + lcol]);
  const unsigned short* ga  = A + (size_t)(m0 + lrow) * D_IN + lcol;
  const unsigned short* gb0 = B + (size_t)(n0 + lrow) * D_IN + lcol;
  const unsigned short* gb1 = B + (size_t)(n0 + lrow + 64) * D_IN + lcol;

  // A fragment: lane L<16 -> row M=L, K = kb..kb+15 (kb = 0 / 16 per lane half)
  const int aoff = (wm * 16 + (lane & 15)) * 40 + (lane >> 4) * 16;
  // B fragments: lane L<16 -> col N=L, K 0..15; L>=16 -> K 16..31
  const int boff = (wn * 64 + (lane & 15)) * 40 + (lane >> 4) * 16;

  for (int kk = 0; kk < D_IN; kk += 32){
    __syncthreads();
    async_copy_b128(a_dst,  ga);
    async_copy_b128(b_dst0, gb0);
    async_copy_b128(b_dst1, gb1);
    if (kk + 32 < D_IN){
      __builtin_prefetch(ga  + 32, 0, 1);
      __builtin_prefetch(gb0 + 32, 0, 1);
      __builtin_prefetch(gb1 + 32, 0, 1);
    }
    ga += 32; gb0 += 32; gb1 += 32;
    wait_async0();
    __syncthreads();

    Frag fa, fb0, fb1, fb2, fb3;
    fa.q[0]  = *(const uint4*)&As[aoff];
    fa.q[1]  = *(const uint4*)&As[aoff + 8];
    fb0.q[0] = *(const uint4*)&Bs[boff];
    fb0.q[1] = *(const uint4*)&Bs[boff + 8];
    fb1.q[0] = *(const uint4*)&Bs[boff + 16 * 40];
    fb1.q[1] = *(const uint4*)&Bs[boff + 16 * 40 + 8];
    fb2.q[0] = *(const uint4*)&Bs[boff + 32 * 40];
    fb2.q[1] = *(const uint4*)&Bs[boff + 32 * 40 + 8];
    fb3.q[0] = *(const uint4*)&Bs[boff + 48 * 40];
    fb3.q[1] = *(const uint4*)&Bs[boff + 48 * 40 + 8];

    acc0 = __builtin_amdgcn_wmma_f32_16x16x32_bf16(false, fa.v, false, fb0.v, (short)0, acc0, false, false);
    acc1 = __builtin_amdgcn_wmma_f32_16x16x32_bf16(false, fa.v, false, fb1.v, (short)0, acc1, false, false);
    acc2 = __builtin_amdgcn_wmma_f32_16x16x32_bf16(false, fa.v, false, fb2.v, (short)0, acc2, false, false);
    acc3 = __builtin_amdgcn_wmma_f32_16x16x32_bf16(false, fa.v, false, fb3.v, (short)0, acc3, false, false);
  }

  // C layout: VGPR r -> M = r (lanes 0-15) / r+8 (lanes 16-31); N = lane&15
  const int mbase = m0 + wm * 16 + (lane >> 4) * 8;
  const int ncb = n0 + wn * 64 + (lane & 15);
  const float b0 = bias[ncb], b1 = bias[ncb + 16], b2 = bias[ncb + 32], b3 = bias[ncb + 48];
  #pragma unroll
  for (int r = 0; r < 8; r++){
    float* row = gx + (size_t)(mbase + r) * N_COMB + ncb;
    row[0]  = acc0[r] + b0;
    row[16] = acc1[r] + b1;
    row[32] = acc2[r] + b2;
    row[48] = acc3[r] + b3;
  }
}

// ------------------------------------------------- stage 2: sequential BiGRU (1 WGP / direction)
// 768 threads; thread g owns gate-row g of W_hh, bf16-packed in 128 VGPRs.
// Dot product uses v_fma_mix_f32_bf16 to consume packed bf16 halves directly (2 VALU / 2 MACs).
__global__ void __launch_bounds__(768, 1)
k_gru(const float* __restrict__ gx,
      const float* __restrict__ Whh_f, const float* __restrict__ bhh_f,
      const float* __restrict__ Whh_b, const float* __restrict__ bhh_b,
      float* __restrict__ hf, float* __restrict__ hb){
  const int dir = blockIdx.x;
  const float* W  = dir ? Whh_b : Whh_f;
  const float* bh = dir ? bhh_b : bhh_f;
  float* hout = dir ? hb : hf;
  const int g = threadIdx.x;

  __shared__ float hsh[H_GRU];
  __shared__ float gh[768];

  unsigned wreg[128];
  {
    const float2* wr = (const float2*)(W + (size_t)g * H_GRU);
    #pragma unroll
    for (int i = 0; i < 128; i++){
      float2 p = wr[i];
      unsigned lo = __float_as_uint(p.x); lo += 0x7fffu + ((lo >> 16) & 1u);
      unsigned hi = __float_as_uint(p.y); hi += 0x7fffu + ((hi >> 16) & 1u);
      wreg[i] = (lo >> 16) | (hi & 0xffff0000u);
    }
  }
  const float bias_g = bh[g];
  if (g < H_GRU) hsh[g] = 0.f;
  __syncthreads();

  for (int t = 0; t < T_LEN; t++){
    const int tt = dir ? (T_LEN - 1 - t) : t;
    float a0 = 0.f, a1 = 0.f, a2 = 0.f, a3 = 0.f;
    const float4* h4 = (const float4*)hsh;          // broadcast reads
    #pragma unroll
    for (int i = 0; i < 64; i++){
      float4 h = h4[i];
      unsigned w0 = wreg[2 * i], w1 = wreg[2 * i + 1];
      // a += bf16_lo(w)*h.x ; a += bf16_hi(w)*h.y ; ...
      asm("v_fma_mix_f32_bf16 %0, %1, %2, %0 op_sel_hi:[1,0,0]"
          : "+v"(a0) : "v"(w0), "v"(h.x));
      asm("v_fma_mix_f32_bf16 %0, %1, %2, %0 op_sel:[1,0,0] op_sel_hi:[1,0,0]"
          : "+v"(a1) : "v"(w0), "v"(h.y));
      asm("v_fma_mix_f32_bf16 %0, %1, %2, %0 op_sel_hi:[1,0,0]"
          : "+v"(a2) : "v"(w1), "v"(h.z));
      asm("v_fma_mix_f32_bf16 %0, %1, %2, %0 op_sel:[1,0,0] op_sel_hi:[1,0,0]"
          : "+v"(a3) : "v"(w1), "v"(h.w));
    }
    gh[g] = bias_g + ((a0 + a1) + (a2 + a3));
    __syncthreads();
    if (g < H_GRU){
      const float* gr = gx + (size_t)tt * N_COMB + dir * 768;
      float r = sigm(gr[g]         + gh[g]);
      float z = sigm(gr[256 + g]   + gh[256 + g]);
      float n = tanhf(gr[512 + g]  + r * gh[512 + g]);
      float hn = (1.f - z) * n + z * hsh[g];
      hsh[g] = hn;
      hout[(size_t)tt * H_GRU + g] = hn;
    }
    __syncthreads();
  }
}

// ------------------------------------------------- stage 3: scorer MLP -> scores[t]
__global__ void __launch_bounds__(128)
k_scorer(const float* __restrict__ hf, const float* __restrict__ hb,
         const float* __restrict__ Ws1, const float* __restrict__ bs1,
         const float* __restrict__ Ws2, const float* __restrict__ bs2,
         float* __restrict__ scores){
  __shared__ float wrow[512];
  const int t = blockIdx.x * 128 + threadIdx.x;
  const float* hfr = hf + (size_t)t * H_GRU;
  const float* hbr = hb + (size_t)t * H_GRU;
  float s = bs2[0];
  for (int j = 0; j < 64; j++){
    for (int i = threadIdx.x; i < 512; i += 128) wrow[i] = Ws1[j * 512 + i];
    __syncthreads();
    float a = bs1[j];
    #pragma unroll 8
    for (int i = 0; i < 256; i++) a += hfr[i] * wrow[i];
    #pragma unroll 8
    for (int i = 0; i < 256; i++) a += hbr[i] * wrow[256 + i];
    s += Ws2[j] * fmaxf(a, 0.f);
    __syncthreads();
  }
  scores[t] = s;
}

// ------------------------------------------------- stage 4: top-8 softmax pool + classifier
__global__ void __launch_bounds__(256)
k_select(const float* __restrict__ scores, const float* __restrict__ features,
         const float* __restrict__ temperature,
         const float* __restrict__ Wc1, const float* __restrict__ bc1,
         const float* __restrict__ Wc2, const float* __restrict__ bc2,
         float* __restrict__ logits){
  __shared__ float pv[256];
  __shared__ int   pi[256];
  __shared__ int   selidx[NF];
  __shared__ float selval[NF];
  __shared__ float w8[NF];
  __shared__ float cr[D_IN];
  __shared__ float hid[256];
  const int tid = threadIdx.x;

  for (int r = 0; r < NF; r++){
    float best = -3.4e38f; int bidx = -1;
    for (int i = tid; i < T_LEN; i += 256){
      bool taken = false;
      for (int q = 0; q < r; q++) if (selidx[q] == i) taken = true;
      float v = scores[i];
      if (!taken && (v > best || (v == best && (unsigned)i < (unsigned)bidx))){ best = v; bidx = i; }
    }
    pv[tid] = best; pi[tid] = bidx;
    __syncthreads();
    if (tid == 0){
      float bb = -3.4e38f; int bi = -1;
      for (int q = 0; q < 256; q++){
        if (pv[q] > bb || (pv[q] == bb && (unsigned)pi[q] < (unsigned)bi)){ bb = pv[q]; bi = pi[q]; }
      }
      selidx[r] = bi; selval[r] = bb;
    }
    __syncthreads();
  }

  if (tid == 0){
    const float tmp = temperature[0];
    float m = -3.4e38f;
    for (int q = 0; q < NF; q++) m = fmaxf(m, selval[q]);
    float den = 0.f;
    for (int q = 0; q < NF; q++){ float e = __expf((selval[q] - m) / tmp); w8[q] = e; den += e; }
    for (int q = 0; q < NF; q++) w8[q] /= den;
  }
  __syncthreads();

  for (int d = tid; d < D_IN; d += 256){
    float a = 0.f;
    #pragma unroll
    for (int q = 0; q < NF; q++) a += w8[q] * features[(size_t)selidx[q] * D_IN + d];
    cr[d] = a;
  }
  __syncthreads();

  {
    float a = bc1[tid];
    const float* w = Wc1 + (size_t)tid * D_IN;
    #pragma unroll 8
    for (int i = 0; i < D_IN; i++) a += cr[i] * w[i];
    hid[tid] = fmaxf(a, 0.f);
  }
  __syncthreads();

  if (tid < NC){
    float a = bc2[tid];
    const float* w = Wc2 + tid * 256;
    for (int j = 0; j < 256; j++) a += hid[j] * w[j];
    logits[tid] = a;
  }
}

// ---------------------------------------------------------------- launcher
extern "C" void kernel_launch(void* const* d_in, const int* in_sizes, int n_in,
                              void* d_out, int out_size, void* d_ws, size_t ws_size,
                              hipStream_t stream){
  const float* features    = (const float*)d_in[0];
  const float* temperature = (const float*)d_in[1];
  const float* W_ih_f = (const float*)d_in[2];
  const float* W_hh_f = (const float*)d_in[3];
  const float* b_ih_f = (const float*)d_in[4];
  const float* b_hh_f = (const float*)d_in[5];
  const float* W_ih_b = (const float*)d_in[6];
  const float* W_hh_b = (const float*)d_in[7];
  const float* b_ih_b = (const float*)d_in[8];
  const float* b_hh_b = (const float*)d_in[9];
  const float* Ws1 = (const float*)d_in[10];
  const float* bs1 = (const float*)d_in[11];
  const float* Ws2 = (const float*)d_in[12];
  const float* bs2 = (const float*)d_in[13];
  const float* Wc1 = (const float*)d_in[14];
  const float* bc1 = (const float*)d_in[15];
  const float* Wc2 = (const float*)d_in[16];
  const float* bc2 = (const float*)d_in[17];
  (void)in_sizes; (void)n_in; (void)out_size; (void)ws_size;

  char* wsb = (char*)d_ws;
  size_t off = 0;
  auto carve = [&](size_t bytes)->char*{
    char* p = wsb + off; off = (off + bytes + 255) & ~(size_t)255; return p;
  };
  unsigned short* Abf = (unsigned short*)carve((size_t)T_LEN * D_IN * 2);     // 64 MB
  unsigned short* Bt  = (unsigned short*)carve((size_t)N_COMB * D_IN * 2);    // 6 MB
  float* biasc = (float*)carve((size_t)N_COMB * 4);
  float* gxbuf = (float*)carve((size_t)T_LEN * N_COMB * 4);                   // 100 MB
  float* hf    = (float*)carve((size_t)T_LEN * H_GRU * 4);                    // 16 MB
  float* hb    = (float*)carve((size_t)T_LEN * H_GRU * 4);                    // 16 MB

  float* out    = (float*)d_out;
  float* logits = out;        // (NC,)
  float* scores = out + NC;   // (T,)

  k_cvt_bf16 <<<4096, 256, 0, stream>>>(features, Abf, T_LEN * D_IN);
  k_pack_w   <<<2048, 256, 0, stream>>>(W_ih_f, W_ih_b, Bt);
  k_pack_bias<<<6,    256, 0, stream>>>(b_ih_f, b_ih_b, biasc);

  dim3 ggrid(N_COMB / 128, T_LEN / 64);
  k_gemm_gx<<<ggrid, 256, 0, stream>>>(Abf, Bt, biasc, gxbuf);

  k_gru   <<<2, 768, 0, stream>>>(gxbuf, W_hh_f, b_hh_f, W_hh_b, b_hh_b, hf, hb);
  k_scorer<<<T_LEN / 128, 128, 0, stream>>>(hf, hb, Ws1, bs1, Ws2, bs2, scores);
  k_select<<<1, 256, 0, stream>>>(scores, features, temperature, Wc1, bc1, Wc2, bc2, logits);
}
